// Decoder_73727408603347
// MI455X (gfx1250) — matrix operand order repaired and checked by reading, not compiled, not run on hardware
//
#include <hip/hip_runtime.h>
#include <stdint.h>

typedef __attribute__((ext_vector_type(16))) __bf16 v16bf;
typedef __attribute__((ext_vector_type(8)))  float  v8f;
typedef unsigned short u16;
typedef unsigned int   u32;

union FragBF { v16bf v; u16 u[16]; };

// ---------------- scalar helpers ----------------
static __device__ __forceinline__ u16 f2bf(float f){
  union { float f; u32 u; } x; x.f = f;
  u32 u = x.u;
  u32 r = (u + 0x7FFFu + ((u >> 16) & 1u)) >> 16;   // round-nearest-even
  return (u16)r;
}
static __device__ __forceinline__ float bf2f(u16 b){
  union { u32 u; float f; } x; x.u = ((u32)b) << 16; return x.f;
}
static __device__ __forceinline__ u32 hashu(u32 x){
  x ^= x >> 17; x *= 0xed5ad4bbu; x ^= x >> 11; x *= 0xac4c1b51u;
  x ^= x >> 15; x *= 0x31848babu; x ^= x >> 14; return x;
}
static __device__ __forceinline__ float keepmask(u32 idx, u32 salt, float p){
  u32 h = hashu(idx * 0x9E3779B1u ^ salt);
  float u = (float)(h >> 8) * (1.0f / 16777216.0f);
  return (u >= p) ? (1.0f / (1.0f - p)) : 0.0f;
}
static __device__ __forceinline__ float gaussn(u32 idx){
  u32 a = hashu(idx ^ 0xA5A5F00Du);
  u32 b = hashu(idx ^ 0x5EEDBEEFu);
  float u1 = ((float)(a >> 8) + 1.0f) * (1.0f / 16777216.0f);
  float u2 = (float)(b >> 8) * (1.0f / 16777216.0f);
  return sqrtf(-2.0f * __logf(u1)) * __cosf(6.28318530718f * u2);
}
static __device__ __forceinline__ float sigm(float x){ return 1.0f / (1.0f + __expf(-x)); }

// packed-B fragment pointer: layout [kt][nt][lane][16 contiguous bf16]
static __device__ __forceinline__ const v16bf* bfrag(const u16* Wp, int kc, int NT, int nt, int lane){
  return reinterpret_cast<const v16bf*>(Wp + ((((size_t)kc * NT + nt) * 32 + lane) << 4));
}
#define WMMA_BF16(A, B, C) \
  __builtin_amdgcn_wmma_f32_16x16x32_bf16(false, (A), false, (B), (short)0, (C), false, false)

// ---------------- setup kernels ----------------
// Pack row-major f32 W[Ks][Ns] (zero-padded to KT*32 x NT*16) into WMMA B-fragment order.
// B lane layout (16-bit, 16x16x32): lane L -> col n0 + L%16 ; K = kt*32 + (j<8 ? (L/16)*8 + j : 16 + (L/16)*8 + j-8)
__global__ void k_pack(u16* dst, const float* src, int Ks, int Ns, int KT, int NT){
  long total = (long)KT * NT * 512;
  for (long e = (long)blockIdx.x * blockDim.x + threadIdx.x; e < total;
       e += (long)gridDim.x * blockDim.x){
    int j    = (int)(e & 15);
    int lane = (int)((e >> 4) & 31);
    long r   = e >> 9;
    int nt   = (int)(r % NT);
    int kt   = (int)(r / NT);
    int n = nt * 16 + (lane & 15);
    int k = kt * 32 + (j < 8 ? ((lane >> 4) << 3) + j : 16 + ((lane >> 4) << 3) + (j - 8));
    float v = (k < Ks && n < Ns) ? src[(size_t)k * Ns + n] : 0.0f;
    dst[e] = f2bf(v);
  }
}

// combined projection weight: cols 0..79 = proj_w, col 80 = gate_w, 81..95 = 0 ; K=1536
__global__ void k_pack_proj(u16* dst, const float* pw, const float* gw){
  long total = (long)48 * 6 * 512;
  for (long e = (long)blockIdx.x * blockDim.x + threadIdx.x; e < total;
       e += (long)gridDim.x * blockDim.x){
    int j    = (int)(e & 15);
    int lane = (int)((e >> 4) & 31);
    long r   = e >> 9;
    int nt   = (int)(r % 6);
    int kt   = (int)(r / 6);
    int n = nt * 16 + (lane & 15);
    int k = kt * 32 + (j < 8 ? ((lane >> 4) << 3) + j : 16 + ((lane >> 4) << 3) + (j - 8));
    float v = 0.0f;
    if (n < 80)       v = pw[(size_t)k * 80 + n];
    else if (n == 80) v = gw[k];
    dst[e] = f2bf(v);
  }
}

__global__ void k_cvt(u16* dst, const float* src, long n){
  for (long i = (long)blockIdx.x * blockDim.x + threadIdx.x; i < n;
       i += (long)gridDim.x * blockDim.x)
    dst[i] = f2bf(src[i]);
}

__global__ void k_init(float* ah, float* ac, float* dc, u16* ahbf, u16* dhbf, u16* ctxbf,
                       float* aw, const float* ah0, const float* ac0, const float* dh0,
                       const float* dc0, const float* aw0, const float* ctx0){
  int b = blockIdx.x;
  for (int n = threadIdx.x; n < 1024; n += blockDim.x){
    size_t ix = (size_t)b * 1024 + n;
    ah[ix] = ah0[n]; ahbf[ix] = f2bf(ah0[n]);
    ac[ix] = ac0[n]; dc[ix] = dc0[n];
    dhbf[ix] = f2bf(dh0[n]);
  }
  for (int n = threadIdx.x; n < 512; n += blockDim.x) ctxbf[(size_t)b * 512 + n] = f2bf(ctx0[n]);
  for (int n = threadIdx.x; n < 200; n += blockDim.x) aw[(size_t)b * 200 + n] = aw0[0];
}

// pm = memory @ wm : A = mem_bf [6400 x 512] row-major -> pm f32 [6400 x 128]
__global__ void __launch_bounds__(256) k_gemm_pm(const u16* Ab, const u16* Wp, float* pm){
  int lane = threadIdx.x & 31, w = threadIdx.x >> 5;
  int id = blockIdx.x * 8 + w;
  int nt = id & 7, mt = id >> 3;
  int mrow  = mt * 16 + (lane & 15);
  int khalf = lane >> 4;
  const u16* arow = Ab + (size_t)mrow * 512;
  v8f acc = {};
  for (int kc = 0; kc < 16; ++kc){
    FragBF a, bm;
    int k0 = kc * 32 + khalf * 8;
    #pragma unroll
    for (int i = 0; i < 8; ++i) a.u[i] = arow[k0 + i];
    #pragma unroll
    for (int i = 0; i < 8; ++i) a.u[8 + i] = arow[k0 + 16 + i];
    bm.v = *bfrag(Wp, kc, 8, nt, lane);
    acc = WMMA_BF16(a.v, bm.v, acc);
  }
  int col = nt * 16 + (lane & 15);
  #pragma unroll
  for (int v = 0; v < 8; ++v){
    int r = mt * 16 + (khalf << 3) + v;
    pm[(size_t)r * 128 + col] = acc[v];
  }
}

// prenet layer 1 over all 800 frames: rows = t*32+b (t=0 -> go frame of zeros)
__global__ void __launch_bounds__(256) k_prenet1(const float* dec, const u16* Wp, u16* h1){
  int lane = threadIdx.x & 31, w = threadIdx.x >> 5;
  int id = blockIdx.x * 8 + w;
  int nt = id & 15, mt = id >> 4;
  int mrow = mt * 16 + (lane & 15);
  int tt = mrow >> 5, bb = mrow & 31;
  int khalf = lane >> 4;
  v8f acc = {};
  for (int kc = 0; kc < 3; ++kc){
    FragBF a, bm;
    #pragma unroll
    for (int j = 0; j < 16; ++j){
      int k = kc * 32 + (j < 8 ? khalf * 8 + j : 16 + khalf * 8 + (j - 8));
      float v = 0.0f;
      if (k < 80 && tt > 0) v = dec[(size_t)bb * 64000 + (size_t)k * 800 + (tt - 1)];
      a.u[j] = f2bf(v);
    }
    bm.v = *bfrag(Wp, kc, 16, nt, lane);
    acc = WMMA_BF16(a.v, bm.v, acc);
  }
  int col = nt * 16 + (lane & 15);
  #pragma unroll
  for (int v = 0; v < 8; ++v){
    int r = mt * 16 + (khalf << 3) + v;
    float x = acc[v]; x = x > 0.f ? x : 0.f;
    x *= keepmask((u32)r * 256u + (u32)col, 0x11111111u, 0.5f);
    h1[(size_t)r * 256 + col] = f2bf(x);
  }
}

// prenet layer 2
__global__ void __launch_bounds__(256) k_prenet2(const u16* h1, const u16* Wp, u16* prebf){
  int lane = threadIdx.x & 31, w = threadIdx.x >> 5;
  int id = blockIdx.x * 8 + w;
  int nt = id & 15, mt = id >> 4;
  int mrow  = mt * 16 + (lane & 15);
  int khalf = lane >> 4;
  const u16* arow = h1 + (size_t)mrow * 256;
  v8f acc = {};
  for (int kc = 0; kc < 8; ++kc){
    FragBF a, bm;
    int k0 = kc * 32 + khalf * 8;
    #pragma unroll
    for (int i = 0; i < 8; ++i) a.u[i] = arow[k0 + i];
    #pragma unroll
    for (int i = 0; i < 8; ++i) a.u[8 + i] = arow[k0 + 16 + i];
    bm.v = *bfrag(Wp, kc, 16, nt, lane);
    acc = WMMA_BF16(a.v, bm.v, acc);
  }
  int col = nt * 16 + (lane & 15);
  #pragma unroll
  for (int v = 0; v < 8; ++v){
    int r = mt * 16 + (khalf << 3) + v;
    float x = acc[v]; x = x > 0.f ? x : 0.f;
    x *= keepmask((u32)r * 256u + (u32)col, 0x22222222u, 0.5f);
    prebf[(size_t)r * 256 + col] = f2bf(x);
  }
}

// ---------------- per-step kernels ----------------
static __device__ __forceinline__ void loadA8_att(u16* dst, int off, int k, int b, int t,
    const u16* prebf, const u16* ctxbf, const u16* ahbf){
  const u16* p;
  if (k < 256)      p = prebf + ((size_t)t * 32 + b) * 256 + k;
  else if (k < 768) p = ctxbf + (size_t)b * 512 + (k - 256);
  else              p = ahbf  + (size_t)b * 1024 + (k - 768);
  #pragma unroll
  for (int i = 0; i < 8; ++i) dst[off + i] = p[i];
}
static __device__ __forceinline__ void loadA8_dec(u16* dst, int off, int k, int b,
    const u16* ahbf, const u16* ctxbf, const u16* dhbf){
  const u16* p;
  if (k < 1024)      p = ahbf + (size_t)b * 1024 + k;
  else if (k < 1536) p = ctxbf + (size_t)b * 512 + (k - 1024);
  else               p = dhbf + (size_t)b * 1024 + (k - 1536);
  #pragma unroll
  for (int i = 0; i < 8; ++i) dst[off + i] = p[i];
}

// attention-LSTM gate GEMM, 8-way split-K: z = [pre_t|ctx|ah] @ [wih;whh]
// 1024 waves: (ks 0..7, mt 0..1, ntB 0..63); each wave -> 4 gate tiles over 7 K-chunks.
// Partials to zpart[((ks*4+g)*32 + b)*1024 + col].
__global__ void __launch_bounds__(128) k_att_gemm(int t, const u16* prebf, const u16* ctxbf,
    const u16* ahbf_in, const u16* Wp, float* zpart){
  int lane = threadIdx.x & 31, w = threadIdx.x >> 5;
  int gw = blockIdx.x * 4 + w;            // 0..1023
  int ntB = gw & 63;
  int r = gw >> 6;                        // 0..15
  int mt = r & 1, ks = r >> 1;            // 4 waves/block share (mt,ks) -> shared A frags
  int brow  = mt * 16 + (lane & 15);
  int khalf = lane >> 4;
  v8f zi = {}, zf = {}, zg = {}, zo = {};
  for (int i = 0; i < 7; ++i){
    int kc = ks * 7 + i;
    FragBF a;
    int k0 = kc * 32 + khalf * 8;
    loadA8_att(a.u, 0, k0, brow, t, prebf, ctxbf, ahbf_in);
    loadA8_att(a.u, 8, k0 + 16, brow, t, prebf, ctxbf, ahbf_in);
    FragBF b0, b1, b2, b3;
    b0.v = *bfrag(Wp, kc, 256,       ntB, lane);
    b1.v = *bfrag(Wp, kc, 256,  64 + ntB, lane);
    b2.v = *bfrag(Wp, kc, 256, 128 + ntB, lane);
    b3.v = *bfrag(Wp, kc, 256, 192 + ntB, lane);
    if (i + 1 < 7) __builtin_prefetch((const void*)bfrag(Wp, kc + 1, 256, ntB, lane), 0, 1);
    zi = WMMA_BF16(a.v, b0.v, zi);
    zf = WMMA_BF16(a.v, b1.v, zf);
    zg = WMMA_BF16(a.v, b2.v, zg);
    zo = WMMA_BF16(a.v, b3.v, zo);
  }
  int col = ntB * 16 + (lane & 15);
  #pragma unroll
  for (int v = 0; v < 8; ++v){
    int b = mt * 16 + (khalf << 3) + v;
    zpart[(((size_t)ks * 4 + 0) * 32 + b) * 1024 + col] = zi[v];
    zpart[(((size_t)ks * 4 + 1) * 32 + b) * 1024 + col] = zf[v];
    zpart[(((size_t)ks * 4 + 2) * 32 + b) * 1024 + col] = zg[v];
    zpart[(((size_t)ks * 4 + 3) * 32 + b) * 1024 + col] = zo[v];
  }
}

// decoder-LSTM gate GEMM, 8-way split-K: z = [ah|ctx|dh] @ [wih;whh]
__global__ void __launch_bounds__(128) k_dec_gemm(const u16* ahbf, const u16* ctxbf,
    const u16* dhbf_in, const u16* Wp, float* zpart){
  int lane = threadIdx.x & 31, w = threadIdx.x >> 5;
  int gw = blockIdx.x * 4 + w;
  int ntB = gw & 63;
  int r = gw >> 6;
  int mt = r & 1, ks = r >> 1;
  int brow  = mt * 16 + (lane & 15);
  int khalf = lane >> 4;
  v8f zi = {}, zf = {}, zg = {}, zo = {};
  for (int i = 0; i < 10; ++i){
    int kc = ks * 10 + i;
    FragBF a;
    int k0 = kc * 32 + khalf * 8;
    loadA8_dec(a.u, 0, k0, brow, ahbf, ctxbf, dhbf_in);
    loadA8_dec(a.u, 8, k0 + 16, brow, ahbf, ctxbf, dhbf_in);
    FragBF b0, b1, b2, b3;
    b0.v = *bfrag(Wp, kc, 256,       ntB, lane);
    b1.v = *bfrag(Wp, kc, 256,  64 + ntB, lane);
    b2.v = *bfrag(Wp, kc, 256, 128 + ntB, lane);
    b3.v = *bfrag(Wp, kc, 256, 192 + ntB, lane);
    if (i + 1 < 10) __builtin_prefetch((const void*)bfrag(Wp, kc + 1, 256, ntB, lane), 0, 1);
    zi = WMMA_BF16(a.v, b0.v, zi);
    zf = WMMA_BF16(a.v, b1.v, zf);
    zg = WMMA_BF16(a.v, b2.v, zg);
    zo = WMMA_BF16(a.v, b3.v, zo);
  }
  int col = ntB * 16 + (lane & 15);
  #pragma unroll
  for (int v = 0; v < 8; ++v){
    int b = mt * 16 + (khalf << 3) + v;
    zpart[(((size_t)ks * 4 + 0) * 32 + b) * 1024 + col] = zi[v];
    zpart[(((size_t)ks * 4 + 1) * 32 + b) * 1024 + col] = zf[v];
    zpart[(((size_t)ks * 4 + 2) * 32 + b) * 1024 + col] = zg[v];
    zpart[(((size_t)ks * 4 + 3) * 32 + b) * 1024 + col] = zo[v];
  }
}

// LSTM cell pointwise: reduce 8 split-K partials per gate, apply gates + dropout.
__global__ void __launch_bounds__(256) k_cell(int t, const float* zpart, const float* bias,
    float* cst, float* hf32, u16* hbf_out, u32 salt){
  int e = blockIdx.x * 256 + threadIdx.x;     // 0..32767
  int b = e >> 10, col = e & 1023;
  float zi = bias[col], zf = bias[1024 + col], zg = bias[2048 + col], zo = bias[3072 + col];
  #pragma unroll
  for (int s = 0; s < 8; ++s){
    size_t base = (((size_t)s * 4) * 32 + b) * 1024 + col;
    zi += zpart[base];
    zf += zpart[base + (size_t)32 * 1024];
    zg += zpart[base + (size_t)64 * 1024];
    zo += zpart[base + (size_t)96 * 1024];
  }
  size_t ix = (size_t)b * 1024 + col;
  float c2 = sigm(zf) * cst[ix] + sigm(zi) * tanhf(zg);
  float h2 = sigm(zo) * tanhf(c2);
  h2 *= keepmask(((u32)t * 32u + (u32)b) * 1024u + (u32)col, salt, 0.1f);
  cst[ix] = c2;
  if (hf32) hf32[ix] = h2;
  hbf_out[ix] = f2bf(h2);
}

// stepwise monotonic attention + context (one block per batch row)
__global__ void __launch_bounds__(256) k_attn(int t, const float* ah, const float* wq,
    const float* pm, const float* v_w, const float* v_b, const float* sb,
    const int* memlen, const u16* membf, const float* aw_in, float* aw_out,
    u16* ctxbf, float* out){
  __shared__ float s_red[256];
  __shared__ float s_pq[128];
  __shared__ float s_p[200];
  __shared__ float s_aw[200];
  int b = blockIdx.x, tid = threadIdx.x;
  { // pq = ah[b] @ wq   (split-K over 2 halves)
    int n = tid & 127, half = tid >> 7;
    const float* arow = ah + (size_t)b * 1024 + half * 512;
    const float* wcol = wq + (size_t)half * 512 * 128 + n;
    float acc = 0.f;
    for (int k = 0; k < 512; ++k) acc += arow[k] * wcol[(size_t)k * 128];
    s_red[tid] = acc;
  }
  __syncthreads();
  if (tid < 128) s_pq[tid] = s_red[tid] + s_red[tid + 128];
  __syncthreads();
  if (tid < 200){
    const float* pmr = pm + ((size_t)b * 200 + tid) * 128;
    float e = sb[0] + v_b[0];
    for (int c = 0; c < 128; ++c) e += tanhf(s_pq[c] + pmr[c]) * v_w[c];
    e += 2.0f * gaussn(((u32)t * 32u + (u32)b) * 200u + (u32)tid);
    s_p[tid] = sigm(e);
  }
  __syncthreads();
  if (tid < 200){
    float awp  = aw_in[(size_t)b * 200 + tid];
    float prev = (tid > 0) ? aw_in[(size_t)b * 200 + tid - 1] * (1.0f - s_p[tid - 1]) : 0.0f;
    float an = awp * s_p[tid] + prev;
    if (tid >= memlen[b]) an = 0.0f;
    aw_out[(size_t)b * 200 + tid] = an;
    s_aw[tid] = an;
    out[(size_t)2073600 + ((size_t)b * 800 + (size_t)t) * 200 + tid] = an;   // aligns
  }
  __syncthreads();
  for (int d = tid; d < 512; d += 256){
    float acc = 0.f;
    for (int j = 0; j < 200; ++j)
      acc += s_aw[j] * bf2f(membf[((size_t)b * 200 + j) * 512 + d]);
    ctxbf[(size_t)b * 512 + d] = f2bf(acc);
  }
}

// mel + gate projection: [dh|ctx] @ [proj_w | gate_w | pad]  (K=1536, N=96)
__global__ void __launch_bounds__(128) k_proj(int t, const u16* dhbf, const u16* ctxbf,
    const u16* Wp, const float* pb, const float* gb, float* out){
  int lane = threadIdx.x & 31, w = threadIdx.x >> 5;
  int gw = blockIdx.x * 4 + w;           // 0..11 : 2 m-tiles x 6 n-tiles
  int mt = gw / 6, nt = gw % 6;
  int brow  = mt * 16 + (lane & 15);
  int khalf = lane >> 4;
  v8f acc = {};
  for (int kc = 0; kc < 48; ++kc){
    FragBF a, bm;
    int k0 = kc * 32 + khalf * 8;
    {
      const u16* p = (k0 < 1024) ? dhbf + (size_t)brow * 1024 + k0
                                 : ctxbf + (size_t)brow * 512 + (k0 - 1024);
      #pragma unroll
      for (int i = 0; i < 8; ++i) a.u[i] = p[i];
    }
    {
      int k1 = k0 + 16;
      const u16* p = (k1 < 1024) ? dhbf + (size_t)brow * 1024 + k1
                                 : ctxbf + (size_t)brow * 512 + (k1 - 1024);
      #pragma unroll
      for (int i = 0; i < 8; ++i) a.u[8 + i] = p[i];
    }
    bm.v = *bfrag(Wp, kc, 6, nt, lane);
    acc = WMMA_BF16(a.v, bm.v, acc);
  }
  int col = nt * 16 + (lane & 15);
  #pragma unroll
  for (int v = 0; v < 8; ++v){
    int b = mt * 16 + (khalf << 3) + v;
    if (col < 80)
      out[(size_t)b * 64000 + (size_t)col * 800 + t] = acc[v] + pb[col];       // mel [B,80,T]
    else if (col == 80)
      out[(size_t)2048000 + (size_t)b * 800 + t] = acc[v] + gb[0];             // gate [B,T]
  }
}

// ---------------- host ----------------
extern "C" void kernel_launch(void* const* d_in, const int* in_sizes, int n_in,
                              void* d_out, int out_size, void* d_ws, size_t ws_size,
                              hipStream_t stream) {
  (void)in_sizes; (void)n_in; (void)out_size;
  const float* memory  = (const float*)d_in[0];
  const float* decin   = (const float*)d_in[1];
  const int*   memlen  = (const int*)  d_in[2];
  const float* pw1     = (const float*)d_in[3];
  const float* pw2     = (const float*)d_in[4];
  const float* att_wih = (const float*)d_in[5];
  const float* att_whh = (const float*)d_in[6];
  const float* att_b   = (const float*)d_in[7];
  const float* wq      = (const float*)d_in[8];
  const float* wm      = (const float*)d_in[9];
  const float* v_w     = (const float*)d_in[10];
  const float* v_b     = (const float*)d_in[11];
  const float* sbias   = (const float*)d_in[12];
  const float* dec_wih = (const float*)d_in[13];
  const float* dec_whh = (const float*)d_in[14];
  const float* dec_b   = (const float*)d_in[15];
  const float* proj_w  = (const float*)d_in[16];
  const float* proj_b  = (const float*)d_in[17];
  const float* gate_w  = (const float*)d_in[18];
  const float* gate_b  = (const float*)d_in[19];
  const float* ah0     = (const float*)d_in[20];
  const float* ac0     = (const float*)d_in[21];
  const float* dh0     = (const float*)d_in[22];
  const float* dc0     = (const float*)d_in[23];
  const float* aw0     = (const float*)d_in[24];
  const float* ctx0    = (const float*)d_in[25];
  float* out = (float*)d_out;

  char* W = (char*)d_ws;
  size_t off = 0;
  auto alloc = [&](size_t bytes){ size_t o = off; off = (off + bytes + 255) & ~(size_t)255; return o; };
  size_t oWATT = alloc((size_t)56 * 256 * 512 * 2);   // [wih;whh] packed bf16
  size_t oWDEC = alloc((size_t)80 * 256 * 512 * 2);
  size_t oWM   = alloc((size_t)16 * 8 * 512 * 2);
  size_t oWP1  = alloc((size_t)3 * 16 * 512 * 2);
  size_t oWP2  = alloc((size_t)8 * 16 * 512 * 2);
  size_t oWPR  = alloc((size_t)48 * 6 * 512 * 2);
  size_t oMEM  = alloc((size_t)6400 * 512 * 2);
  size_t oPRE  = alloc((size_t)800 * 32 * 256 * 2);
  size_t oH1   = alloc((size_t)25600 * 256 * 2);
  size_t oPM   = alloc((size_t)6400 * 128 * 4);
  size_t oZP   = alloc((size_t)8 * 4 * 32 * 1024 * 4);  // split-K gate partials
  size_t oAH   = alloc((size_t)32768 * 4);
  size_t oAC   = alloc((size_t)32768 * 4);
  size_t oDC   = alloc((size_t)32768 * 4);
  size_t oAHB  = alloc((size_t)2 * 32768 * 2);
  size_t oDHB  = alloc((size_t)2 * 32768 * 2);
  size_t oCTX  = alloc((size_t)16384 * 2);
  size_t oAW   = alloc((size_t)2 * 6400 * 4);
  if (off > ws_size) return;  // not enough scratch

  u16*   WATT = (u16*)(W + oWATT);
  u16*   WDEC = (u16*)(W + oWDEC);
  u16*   WM   = (u16*)(W + oWM);
  u16*   WP1  = (u16*)(W + oWP1);
  u16*   WP2  = (u16*)(W + oWP2);
  u16*   WPR  = (u16*)(W + oWPR);
  u16*   MEMB = (u16*)(W + oMEM);
  u16*   PREB = (u16*)(W + oPRE);
  u16*   H1B  = (u16*)(W + oH1);
  float* PM   = (float*)(W + oPM);
  float* ZP   = (float*)(W + oZP);
  float* AH   = (float*)(W + oAH);
  float* AC   = (float*)(W + oAC);
  float* DC   = (float*)(W + oDC);
  u16*   AHB  = (u16*)(W + oAHB);
  u16*   DHB  = (u16*)(W + oDHB);
  u16*   CTXB = (u16*)(W + oCTX);
  float* AW   = (float*)(W + oAW);

  // ---- one-time setup (all on stream; deterministic each call) ----
  k_pack<<<2048, 256, 0, stream>>>(WATT,                        att_wih,  768, 4096, 24, 256);
  k_pack<<<2048, 256, 0, stream>>>(WATT + (size_t)24*256*512,   att_whh, 1024, 4096, 32, 256);
  k_pack<<<2048, 256, 0, stream>>>(WDEC,                        dec_wih, 1536, 4096, 48, 256);
  k_pack<<<2048, 256, 0, stream>>>(WDEC + (size_t)48*256*512,   dec_whh, 1024, 4096, 32, 256);
  k_pack<<<256,  256, 0, stream>>>(WM,  wm,  512, 128, 16, 8);
  k_pack<<<64,   256, 0, stream>>>(WP1, pw1,  80, 256,  3, 16);
  k_pack<<<128,  256, 0, stream>>>(WP2, pw2, 256, 256,  8, 16);
  k_pack_proj<<<256, 256, 0, stream>>>(WPR, proj_w, gate_w);
  k_cvt<<<2048, 256, 0, stream>>>(MEMB, memory, (long)6400 * 512);
  k_init<<<32, 256, 0, stream>>>(AH, AC, DC, AHB, DHB, CTXB, AW, ah0, ac0, dh0, dc0, aw0, ctx0);
  k_gemm_pm<<<400, 256, 0, stream>>>(MEMB, WM, PM);
  k_prenet1<<<3200, 256, 0, stream>>>(decin, WP1, H1B);
  k_prenet2<<<3200, 256, 0, stream>>>(H1B, WP2, PREB);

  // ---- sequential decoder scan (double-buffered state) ----
  for (int t = 0; t < 800; ++t){
    int cur = t & 1, nxt = cur ^ 1;
    // attention LSTM: split-K GEMM (1024 waves) + cell reduce
    k_att_gemm<<<256, 128, 0, stream>>>(t, PREB, CTXB, AHB + (size_t)cur * 32768, WATT, ZP);
    k_cell<<<128, 256, 0, stream>>>(t, ZP, att_b, AC, AH, AHB + (size_t)nxt * 32768, 0x33333333u);
    // monotonic attention + context
    k_attn<<<32, 256, 0, stream>>>(t, AH, wq, PM, v_w, v_b, sbias, memlen, MEMB,
                                   AW + (size_t)cur * 6400, AW + (size_t)nxt * 6400,
                                   CTXB, out);
    // decoder LSTM: split-K GEMM + cell reduce
    k_dec_gemm<<<256, 128, 0, stream>>>(AHB + (size_t)nxt * 32768, CTXB,
                                        DHB + (size_t)cur * 32768, WDEC, ZP);
    k_cell<<<128, 256, 0, stream>>>(t, ZP, dec_b, DC, nullptr,
                                    DHB + (size_t)nxt * 32768, 0x44444444u);
    // mel + gate projection
    k_proj<<<3, 128, 0, stream>>>(t, DHB + (size_t)nxt * 32768, CTXB, WPR,
                                  proj_b, gate_b, out);
  }
}
